// ResidualVectorQuantizer_27264452395731
// MI455X (gfx1250) — matrix operand order repaired and checked by reading, not compiled, not run on hardware
//
#include <hip/hip_runtime.h>

// ---- WMMA vector types (gfx1250, wave32) ----
typedef __attribute__((ext_vector_type(16))) __bf16 v16bf;
typedef __attribute__((ext_vector_type(16))) unsigned short v16us;
typedef __attribute__((ext_vector_type(8)))  float v8f;

static constexpr int DDIM  = 64;    // vector dimension
static constexpr int CODES = 256;   // codes per stage
static constexpr int RPB   = 128;   // rows per block (8 waves x 16 rows)
static constexpr int CBS   = 72;    // padded LDS stride (bf16 elems) -> bank spread

__device__ __forceinline__ unsigned short f2bf(float f) {
  union { float f; unsigned u; } a; a.f = f;
  // round-to-nearest-even fp32 -> bf16
  return (unsigned short)((a.u + 0x7FFFu + ((a.u >> 16) & 1u)) >> 16);
}

union VU { v16us v; uint4 q[2]; unsigned short s[16]; };

__global__ __launch_bounds__(256)
void rvq_wmma_kernel(const float* __restrict__ x, const float* __restrict__ cbs,
                     float* __restrict__ xq, float* __restrict__ loss,
                     int* __restrict__ idx_out, int nRows, int nStages)
{
  __shared__ __align__(16) unsigned short cb[CODES][CBS]; // bf16 codebook
  __shared__ float cnorm[CODES];
  __shared__ int   selIdx[RPB];
  __shared__ float redbuf[8];

  const int tid  = threadIdx.x;
  const int wave = tid >> 5;
  const int lane = tid & 31;
  const int half = lane >> 4;       // 0: lanes 0-15, 1: lanes 16-31
  const int l16  = lane & 15;
  const long long gr  = (long long)blockIdx.x * RPB + wave * 16 + l16;
  const bool valid    = (gr < (long long)nRows);
  const long long grc = valid ? gr : (long long)(nRows - 1); // clamp for safe loads

  // Residual lives in registers, laid out exactly as the WMMA A fragment:
  // lane (l16, half) owns row l16, chunk t covers cols t*16 + half*8 .. +7.
  float res[4][8];
  {
    const float* xr = x + grc * DDIM;
    #pragma unroll
    for (int t = 0; t < 4; ++t) {
      const float4* p = (const float4*)(xr + t * 16 + half * 8);
      float4 a = p[0], b = p[1];
      res[t][0]=a.x; res[t][1]=a.y; res[t][2]=a.z; res[t][3]=a.w;
      res[t][4]=b.x; res[t][5]=b.y; res[t][6]=b.z; res[t][7]=b.w;
    }
  }

  float lsum = 0.0f;

  for (int stage = 0; stage < nStages; ++stage) {
    __syncthreads(); // prior stage done reading cb
    // Stage codebook -> LDS bf16, plus exact f32 norms (one code per thread)
    {
      const float* crow = cbs + ((long long)stage * CODES + tid) * DDIM;
      float nrm = 0.f;
      #pragma unroll
      for (int c = 0; c < DDIM; c += 4) {
        float4 v = *(const float4*)(crow + c);
        cb[tid][c+0] = f2bf(v.x); cb[tid][c+1] = f2bf(v.y);
        cb[tid][c+2] = f2bf(v.z); cb[tid][c+3] = f2bf(v.w);
        nrm += v.x*v.x + v.y*v.y + v.z*v.z + v.w*v.w;
      }
      cnorm[tid] = nrm;
    }
    __syncthreads();

    // Build A fragments (16x32 bf16 each) from the residual registers.
    // Vector elems 0..7 = K(lo 8), elems 8..15 = K(+16) per ISA layout.
    VU ua0, ua1;
    #pragma unroll
    for (int e = 0; e < 8; ++e) {
      ua0.s[e]     = f2bf(res[0][e]);
      ua0.s[8 + e] = f2bf(res[1][e]);
      ua1.s[e]     = f2bf(res[2][e]);
      ua1.s[8 + e] = f2bf(res[3][e]);
    }
    const v16bf A0 = __builtin_bit_cast(v16bf, ua0.v);
    const v16bf A1 = __builtin_bit_cast(v16bf, ua1.v);

    float minv[8]; int mini[8];
    #pragma unroll
    for (int j = 0; j < 8; ++j) { minv[j] = 3.4e38f; mini[j] = 0; }

    // 16 N-tiles of 16 codes; score = ||c||^2 - 2 * (r . c).
    // Two INDEPENDENT accumulators per tile (no D->C serialization between
    // the two WMMAs), and 2x unroll so the next tile's loads/WMMAs fill the
    // WMMA->VALU hazard slots instead of v_nops.
    #pragma unroll 2
    for (int nt = 0; nt < 16; ++nt) {
      const int cr = nt * 16 + l16; // this lane's code row for the B fragment
      VU ub0, ub1;
      const uint4* b0p = (const uint4*)&cb[cr][half * 8];
      const uint4* b1p = (const uint4*)&cb[cr][32 + half * 8];
      ub0.q[0] = b0p[0]; ub0.q[1] = b0p[2];
      ub1.q[0] = b1p[0]; ub1.q[1] = b1p[2];
      v8f c0 = {}, c1 = {};
      c0 = __builtin_amdgcn_wmma_f32_16x16x32_bf16(
              false, A0, false, __builtin_bit_cast(v16bf, ub0.v),
              (short)0, c0, false, false);
      c1 = __builtin_amdgcn_wmma_f32_16x16x32_bf16(
              false, A1, false, __builtin_bit_cast(v16bf, ub1.v),
              (short)0, c1, false, false);
      const int   n  = nt * 16 + l16;       // code index for this lane's column
      const float cn = cnorm[n];
      #pragma unroll
      for (int j = 0; j < 8; ++j) {         // VGPR j -> row j + 8*half
        float s = cn - 2.0f * (c0[j] + c1[j]);
        if (s < minv[j]) { minv[j] = s; mini[j] = n; }
      }
    }

    // Per-row argmin across the 16 lanes of each half (lower index wins ties)
    #pragma unroll
    for (int j = 0; j < 8; ++j) {
      float v = minv[j]; int i = mini[j];
      #pragma unroll
      for (int off = 1; off < 16; off <<= 1) {
        float ov = __shfl_xor(v, off, 32);
        int   oi = __shfl_xor(i, off, 32);
        if (ov < v || (ov == v && oi < i)) { v = ov; i = oi; }
      }
      if (l16 == j) {
        const int lr = wave * 16 + j + 8 * half;
        selIdx[lr] = i;
        const long long g = (long long)blockIdx.x * RPB + lr;
        if (g < (long long)nRows) idx_out[g * nStages + stage] = i;
      }
    }
    __syncthreads();

    // Exact fp32 residual update from the global codebook; accumulate loss
    {
      const int code = selIdx[wave * 16 + l16];
      const float* crow = cbs + ((long long)stage * CODES + code) * DDIM;
      float part = 0.f;
      #pragma unroll
      for (int t = 0; t < 4; ++t) {
        const float4* p = (const float4*)(crow + t * 16 + half * 8);
        float4 a = p[0], b = p[1];
        res[t][0]-=a.x; res[t][1]-=a.y; res[t][2]-=a.z; res[t][3]-=a.w;
        res[t][4]-=b.x; res[t][5]-=b.y; res[t][6]-=b.z; res[t][7]-=b.w;
        #pragma unroll
        for (int e = 0; e < 8; ++e) part += res[t][e] * res[t][e];
      }
      if (valid) lsum += part;
    }
  }

  // x_q = x - final residual
  if (valid) {
    const float* xr = x  + gr * DDIM;
    float*       o  = xq + gr * DDIM;
    #pragma unroll
    for (int t = 0; t < 4; ++t) {
      const float4* p = (const float4*)(xr + t * 16 + half * 8);
      float4 a = p[0], b = p[1];
      float4 oa, ob;
      oa.x = a.x - res[t][0]; oa.y = a.y - res[t][1];
      oa.z = a.z - res[t][2]; oa.w = a.w - res[t][3];
      ob.x = b.x - res[t][4]; ob.y = b.y - res[t][5];
      ob.z = b.z - res[t][6]; ob.w = b.w - res[t][7];
      ((float4*)(o + t * 16 + half * 8))[0] = oa;
      ((float4*)(o + t * 16 + half * 8))[1] = ob;
    }
  }

  // Loss: loss_s = (1+beta)*mean(residual_after^2); mean over stages.
  #pragma unroll
  for (int off = 16; off >= 1; off >>= 1) lsum += __shfl_xor(lsum, off, 32);
  if (lane == 0) redbuf[wave] = lsum;
  __syncthreads();
  if (tid == 0) {
    float t = 0.f;
    #pragma unroll
    for (int w = 0; w < 8; ++w) t += redbuf[w];
    const float scale = 1.25f / ((float)nStages * (float)nRows * (float)DDIM);
    atomicAdd(loss, t * scale);
  }
}

extern "C" void kernel_launch(void* const* d_in, const int* in_sizes, int n_in,
                              void* d_out, int out_size, void* d_ws, size_t ws_size,
                              hipStream_t stream) {
  const float* x   = (const float*)d_in[0];
  const float* cbs = (const float*)d_in[1];
  const int nRows   = in_sizes[0] / DDIM;
  const int nStages = in_sizes[1] / (CODES * DDIM);

  float* xq      = (float*)d_out;
  float* loss    = xq + (size_t)nRows * DDIM;
  int*   idx_out = (int*)(loss + 1);

  hipMemsetAsync(loss, 0, sizeof(float), stream); // loss accumulator -> 0

  const int blocks = (nRows + RPB - 1) / RPB;
  rvq_wmma_kernel<<<blocks, 256, 0, stream>>>(x, cbs, xq, loss, idx_out,
                                              nRows, nStages);
}
